// SelfAttentionV3_22273700397196
// MI455X (gfx1250) — compile-verified, matrix operand order checked by reading, not run on hardware
//
#include <hip/hip_runtime.h>
#include <hip/hip_bf16.h>

typedef __attribute__((ext_vector_type(16))) __bf16 v16bf;
typedef __attribute__((ext_vector_type(8)))  float  v8f;
typedef unsigned short u16;
typedef unsigned int   u32;
typedef unsigned long long u64;

// Problem constants (match reference)
constexpr int kB  = 4;
constexpr int kS  = 2048;
constexpr int kD  = 1024;
constexpr int kM  = kB * kS;        // 8192 token rows
constexpr int kN3 = 3 * kD;         // 3072
constexpr float kScale = 0.03125f;  // 1/sqrt(1024)

// ---------- bf16 helpers ----------
__device__ __forceinline__ u16 f32_to_bf16(float f) {
  u32 u = __float_as_uint(f);
  u32 r = u + 0x7FFFu + ((u >> 16) & 1u);   // round-to-nearest-even
  return (u16)(r >> 16);
}

union BF16x16 { v16bf v; u16 u[16]; uint4 q[2]; };

// 16x32 bf16 WMMA A/B operand tile loader from a row-major (K-contiguous)
// matrix with row stride `ld` elements. Lanes 0-15: row=lane, K={0..7,16..23};
// lanes 16-31: row=lane-16, K={8..15,24..31}.
__device__ __forceinline__ v16bf load_ab16(const u16* __restrict__ base, int ld) {
  int lane = threadIdx.x & 31;
  const u16* p = base + (size_t)(lane & 15) * ld + ((lane >> 4) << 3);
  BF16x16 t;
  t.q[0] = *(const uint4*)(p);
  t.q[1] = *(const uint4*)(p + 16);
  return t.v;
}

__device__ __forceinline__ v8f wmma_bf16(v16bf a, v16bf b, v8f c) {
  return __builtin_amdgcn_wmma_f32_16x16x32_bf16(
      false, a, false, b, (short)0, c, false, false);
}

__device__ __forceinline__ v8f zero8() {
  return v8f{0.f,0.f,0.f,0.f,0.f,0.f,0.f,0.f};
}

// ---------- CDNA5 async global->LDS copy (ASYNCcnt-tracked) ----------
__device__ __forceinline__ void async_g2l_b128(u32 lds_off, const void* gaddr) {
  asm volatile("global_load_async_to_lds_b128 %0, %1, off"
               :: "v"(lds_off), "v"((u64)(uintptr_t)gaddr)
               : "memory");
}
__device__ __forceinline__ void wait_async_le16() {
  asm volatile("s_wait_asynccnt 0x10" ::: "memory");
}
__device__ __forceinline__ void wait_async_0() {
  asm volatile("s_wait_asynccnt 0x0" ::: "memory");
}

// ---------- precision / layout preparation kernels ----------
__global__ void cvt_bf16_kernel(const float* __restrict__ src,
                                u16* __restrict__ dst, int n) {
  int i = blockIdx.x * blockDim.x + threadIdx.x;
  if (i < n) dst[i] = f32_to_bf16(src[i]);
}

// dst[c*R + r] = bf16(src[r*C + c])   (weights -> B^T layout, K-contiguous)
__global__ void transpose_bf16_kernel(const float* __restrict__ src,
                                      u16* __restrict__ dst, int R, int C) {
  int i = blockIdx.x * blockDim.x + threadIdx.x;
  if (i < R * C) {
    int r = i / C, c = i % C;
    dst[(size_t)c * R + r] = f32_to_bf16(src[i]);
  }
}

// ---------- fused QKV projection: qkv = x @ W_qkv + b ----------
// Wave tile 32(M) x 64(N), double-buffered K chunks of 32.
__global__ __launch_bounds__(256)
void gemm_qkv_kernel(const u16* __restrict__ xb, const u16* __restrict__ wt,
                     const float* __restrict__ bias,
                     u16* __restrict__ qb, u16* __restrict__ kb,
                     u16* __restrict__ vT) {
  const int wave = threadIdx.x >> 5;
  const int lane = threadIdx.x & 31;
  const int m0 = blockIdx.x * 32;
  const int n0 = blockIdx.y * 512 + wave * 64;

  v8f acc[2][4];
#pragma unroll
  for (int mi = 0; mi < 2; ++mi)
#pragma unroll
    for (int t = 0; t < 4; ++t) acc[mi][t] = zero8();

  v16bf A0[2], B0[4], A1[2], B1[4];
#pragma unroll
  for (int mi = 0; mi < 2; ++mi)
    A0[mi] = load_ab16(xb + (size_t)(m0 + mi * 16) * kD, kD);
#pragma unroll
  for (int t = 0; t < 4; ++t)
    B0[t] = load_ab16(wt + (size_t)(n0 + t * 16) * kD, kD);

  for (int kk = 0; kk < kD; kk += 64) {
    // prefetch chunk kk+32 while computing chunk kk
#pragma unroll
    for (int mi = 0; mi < 2; ++mi)
      A1[mi] = load_ab16(xb + (size_t)(m0 + mi * 16) * kD + kk + 32, kD);
#pragma unroll
    for (int t = 0; t < 4; ++t)
      B1[t] = load_ab16(wt + (size_t)(n0 + t * 16) * kD + kk + 32, kD);
#pragma unroll
    for (int mi = 0; mi < 2; ++mi)
#pragma unroll
      for (int t = 0; t < 4; ++t)
        acc[mi][t] = wmma_bf16(A0[mi], B0[t], acc[mi][t]);

    if (kk + 64 < kD) {
#pragma unroll
      for (int mi = 0; mi < 2; ++mi)
        A0[mi] = load_ab16(xb + (size_t)(m0 + mi * 16) * kD + kk + 64, kD);
#pragma unroll
      for (int t = 0; t < 4; ++t)
        B0[t] = load_ab16(wt + (size_t)(n0 + t * 16) * kD + kk + 64, kD);
    }
#pragma unroll
    for (int mi = 0; mi < 2; ++mi)
#pragma unroll
      for (int t = 0; t < 4; ++t)
        acc[mi][t] = wmma_bf16(A1[mi], B1[t], acc[mi][t]);
  }

  const int col16 = lane & 15;
  const int hi = lane >> 4;
#pragma unroll
  for (int mi = 0; mi < 2; ++mi) {
#pragma unroll
    for (int t = 0; t < 4; ++t) {
      const int e = n0 + t * 16 + col16;
      const float bv = bias[e];
#pragma unroll
      for (int r = 0; r < 8; ++r) {
        const int tok = m0 + mi * 16 + r + hi * 8;
        const u16 h = f32_to_bf16(acc[mi][t][r] + bv);
        if (e < kD) {
          qb[(size_t)tok * kD + e] = h;
        } else if (e < 2 * kD) {
          kb[(size_t)tok * kD + (e - kD)] = h;
        } else {
          const int bidx = tok >> 11;
          const int s    = tok & (kS - 1);
          vT[((size_t)bidx * kD + (e - 2 * kD)) * kS + s] = h;
        }
      }
    }
  }
}

// ---------- streaming-softmax attention ----------
// WG = 8 waves / 16 query rows. K tiles (32 keys x 1024 d, 64KB) are staged
// into LDS with double-buffered async copies; waves partition D (128 each)
// for Q.K^T partials (ds_add_f32 reduce) and own 128 output cols of P.V.
constexpr int kKLd = 1032;           // padded LDS row stride (elements)
constexpr int kNT  = kS / 32;        // 64 key tiles

__device__ __forceinline__ void stage_ktile(const u16* __restrict__ gbase,
                                            u32 lds_base) {
  const int t = threadIdx.x;
#pragma unroll
  for (int i = 0; i < 16; ++i) {
    const int e   = i * 256 + t;      // 16B-chunk index, 0..4095
    const int row = e >> 7;           // 128 chunks per 2048B row
    const int col = e & 127;
    const char* g = (const char*)gbase + (size_t)row * 2048 + (size_t)col * 16;
    const u32 l = lds_base + (u32)row * (kKLd * 2) + (u32)col * 16;
    async_g2l_b128(l, g);
  }
}

__global__ __launch_bounds__(256)
void attn_kernel(const u16* __restrict__ q, const u16* __restrict__ k,
                 const u16* __restrict__ vT, u16* __restrict__ ao) {
  __shared__ u16   kl[2][32 * kKLd];  // 2 x 66KB staged K tiles
  __shared__ float sc[16 * 32];       // reduced score tile
  __shared__ u16   pb[16 * 32];       // probabilities bf16

  const int wave  = threadIdx.x >> 5;
  const int lane  = threadIdx.x & 31;
  const int col16 = lane & 15;
  const int hi    = lane >> 4;
  const int tok0  = blockIdx.x * 16;
  const int b     = tok0 >> 11;
  const int dsl   = wave * 128;

  const u32 kl_off[2] = { (u32)(uintptr_t)&kl[0][0], (u32)(uintptr_t)&kl[1][0] };

  // Preload Q A-operand tiles (4 chunks of K=32) for this wave's D slice.
  v16bf QA[4];
#pragma unroll
  for (int c = 0; c < 4; ++c)
    QA[c] = load_ab16(q + (size_t)tok0 * kD + dsl + c * 32, kD);

  v8f acc[8];
#pragma unroll
  for (int t = 0; t < 8; ++t) acc[t] = zero8();

  float m_run[8], l_run[8];
#pragma unroll
  for (int r = 0; r < 8; ++r) { m_run[r] = -1e30f; l_run[r] = 0.f; }

  // stage tile 0
  stage_ktile(k + (size_t)(b * kS) * kD, kl_off[0]);
  int cur = 0;

  for (int kt = 0; kt < kNT; ++kt) {
    // issue async copies for the next tile into the alternate buffer
    if (kt + 1 < kNT) {
      stage_ktile(k + (size_t)(b * kS + (kt + 1) * 32) * kD, kl_off[cur ^ 1]);
      wait_async_le16();   // drain tile kt's copies (in-order completion)
    } else {
      wait_async_0();
    }
    sc[threadIdx.x] = 0.f;
    sc[threadIdx.x + 256] = 0.f;
    __syncthreads();       // tile kt fully in LDS, sc zeroed

    // ---- partial scores over this wave's 128-wide D slice (from LDS) ----
    const u16* klp = (const u16*)&kl[cur][0];
    v8f c0 = zero8(), c1 = zero8();
#pragma unroll
    for (int c = 0; c < 4; ++c) {
      v16bf b0 = load_ab16(klp + dsl + c * 32, kKLd);               // keys 0..15
      v16bf b1 = load_ab16(klp + 16 * kKLd + dsl + c * 32, kKLd);   // keys 16..31
      c0 = wmma_bf16(QA[c], b0, c0);
      c1 = wmma_bf16(QA[c], b1, c1);
    }

    // ---- hoist V tile loads (independent of scores) ----
    v16bf VB[8];
#pragma unroll
    for (int t = 0; t < 8; ++t) {
      const u16* vb = vT + ((size_t)b * kD + dsl + t * 16) * kS + kt * 32;
      VB[t] = load_ab16(vb, kS);
    }

#pragma unroll
    for (int r = 0; r < 8; ++r) {
      const int row = r + hi * 8;
      atomicAdd(&sc[row * 32 + col16], c0[r]);
      atomicAdd(&sc[row * 32 + 16 + col16], c1[r]);
    }
    __syncthreads();

    // ---- online softmax (each wave keeps identical running stats) ----
    float s0[8], s1[8], alpha[8];
#pragma unroll
    for (int r = 0; r < 8; ++r) {
      const int row = r + hi * 8;
      s0[r] = sc[row * 32 + col16] * kScale;
      s1[r] = sc[row * 32 + 16 + col16] * kScale;
      float mx = fmaxf(s0[r], s1[r]);
#pragma unroll
      for (int off = 1; off < 16; off <<= 1)
        mx = fmaxf(mx, __shfl_xor(mx, off, 32));
      const float m_new = fmaxf(m_run[r], mx);
      alpha[r] = __expf(m_run[r] - m_new);
      s0[r] = __expf(s0[r] - m_new);
      s1[r] = __expf(s1[r] - m_new);
      float rs = s0[r] + s1[r];
#pragma unroll
      for (int off = 1; off < 16; off <<= 1)
        rs += __shfl_xor(rs, off, 32);
      l_run[r] = l_run[r] * alpha[r] + rs;
      m_run[r] = m_new;
    }
#pragma unroll
    for (int t = 0; t < 8; ++t)
#pragma unroll
      for (int r = 0; r < 8; ++r)
        acc[t][r] *= alpha[r];

    // ---- P (C layout) -> LDS -> A layout ----
    if (wave == 0) {
#pragma unroll
      for (int r = 0; r < 8; ++r) {
        const int row = r + hi * 8;
        pb[row * 32 + col16]      = f32_to_bf16(s0[r]);
        pb[row * 32 + 16 + col16] = f32_to_bf16(s1[r]);
      }
    }
    __syncthreads();
    v16bf PA = load_ab16((const u16*)pb, 32);

    // ---- acc += P @ V for this wave's 128 output columns ----
#pragma unroll
    for (int t = 0; t < 8; ++t)
      acc[t] = wmma_bf16(PA, VB[t], acc[t]);

    __syncthreads();
    cur ^= 1;
  }

  // ---- normalize and store bf16 attention output ----
#pragma unroll
  for (int t = 0; t < 8; ++t) {
#pragma unroll
    for (int r = 0; r < 8; ++r) {
      const int tok = tok0 + r + hi * 8;
      const int col = dsl + t * 16 + col16;
      ao[(size_t)tok * kD + col] = f32_to_bf16(acc[t][r] / l_run[r]);
    }
  }
}

// ---------- output projection: out = attn @ W_out + b (f32 result) ----------
__global__ __launch_bounds__(256)
void gemm_out_kernel(const u16* __restrict__ ab, const u16* __restrict__ wt,
                     const float* __restrict__ bias, float* __restrict__ out) {
  const int wave = threadIdx.x >> 5;
  const int lane = threadIdx.x & 31;
  const int m0 = blockIdx.x * 32;
  const int n0 = blockIdx.y * 512 + wave * 64;

  v8f acc[2][4];
#pragma unroll
  for (int mi = 0; mi < 2; ++mi)
#pragma unroll
    for (int t = 0; t < 4; ++t) acc[mi][t] = zero8();

  v16bf A0[2], B0[4], A1[2], B1[4];
#pragma unroll
  for (int mi = 0; mi < 2; ++mi)
    A0[mi] = load_ab16(ab + (size_t)(m0 + mi * 16) * kD, kD);
#pragma unroll
  for (int t = 0; t < 4; ++t)
    B0[t] = load_ab16(wt + (size_t)(n0 + t * 16) * kD, kD);

  for (int kk = 0; kk < kD; kk += 64) {
#pragma unroll
    for (int mi = 0; mi < 2; ++mi)
      A1[mi] = load_ab16(ab + (size_t)(m0 + mi * 16) * kD + kk + 32, kD);
#pragma unroll
    for (int t = 0; t < 4; ++t)
      B1[t] = load_ab16(wt + (size_t)(n0 + t * 16) * kD + kk + 32, kD);
#pragma unroll
    for (int mi = 0; mi < 2; ++mi)
#pragma unroll
      for (int t = 0; t < 4; ++t)
        acc[mi][t] = wmma_bf16(A0[mi], B0[t], acc[mi][t]);

    if (kk + 64 < kD) {
#pragma unroll
      for (int mi = 0; mi < 2; ++mi)
        A0[mi] = load_ab16(ab + (size_t)(m0 + mi * 16) * kD + kk + 64, kD);
#pragma unroll
      for (int t = 0; t < 4; ++t)
        B0[t] = load_ab16(wt + (size_t)(n0 + t * 16) * kD + kk + 64, kD);
    }
#pragma unroll
    for (int mi = 0; mi < 2; ++mi)
#pragma unroll
      for (int t = 0; t < 4; ++t)
        acc[mi][t] = wmma_bf16(A1[mi], B1[t], acc[mi][t]);
  }

  const int col16 = lane & 15;
  const int hi = lane >> 4;
#pragma unroll
  for (int mi = 0; mi < 2; ++mi) {
#pragma unroll
    for (int t = 0; t < 4; ++t) {
      const int e = n0 + t * 16 + col16;
      const float bv = bias[e];
#pragma unroll
      for (int r = 0; r < 8; ++r) {
        const int tok = m0 + mi * 16 + r + hi * 8;
        out[(size_t)tok * kD + e] = acc[mi][t][r] + bv;
      }
    }
  }
}

extern "C" void kernel_launch(void* const* d_in, const int* in_sizes, int n_in,
                              void* d_out, int out_size, void* d_ws, size_t ws_size,
                              hipStream_t stream) {
  const float* x     = (const float*)d_in[0];
  const float* W_qkv = (const float*)d_in[1];
  const float* b_qkv = (const float*)d_in[2];
  const float* W_out = (const float*)d_in[3];
  const float* b_out = (const float*)d_in[4];
  float* out = (float*)d_out;

  char* ws = (char*)d_ws;
  u16* xb     = (u16*)ws;  ws += (size_t)kM  * kD * 2;
  u16* wqkv_t = (u16*)ws;  ws += (size_t)kN3 * kD * 2;
  u16* wout_t = (u16*)ws;  ws += (size_t)kD  * kD * 2;
  u16* qb     = (u16*)ws;  ws += (size_t)kM  * kD * 2;
  u16* kb     = (u16*)ws;  ws += (size_t)kM  * kD * 2;
  u16* vT     = (u16*)ws;  ws += (size_t)kM  * kD * 2;   // [B][D][S]
  u16* attb   = (u16*)ws;  ws += (size_t)kM  * kD * 2;

  const int nx = kM * kD;
  cvt_bf16_kernel<<<(nx + 255) / 256, 256, 0, stream>>>(x, xb, nx);
  transpose_bf16_kernel<<<(kD * kN3 + 255) / 256, 256, 0, stream>>>(
      W_qkv, wqkv_t, kD, kN3);
  transpose_bf16_kernel<<<(kD * kD + 255) / 256, 256, 0, stream>>>(
      W_out, wout_t, kD, kD);

  gemm_qkv_kernel<<<dim3(kM / 32, kN3 / 512), 256, 0, stream>>>(
      xb, wqkv_t, b_qkv, qb, kb, vT);

  attn_kernel<<<kM / 16, 256, 0, stream>>>(qb, kb, vT, attb);

  gemm_out_kernel<<<dim3(kM / 32, kD / 512), 256, 0, stream>>>(
      attb, wout_t, b_out, out);
}
